// Attention_74191265071149
// MI455X (gfx1250) — matrix operand order repaired and checked by reading, not compiled, not run on hardware
//
#include <hip/hip_runtime.h>
#include <hip/hip_bf16.h>

// ---------------------------------------------------------------------------
// CDNA5 (gfx1250, wave32) fused window-attention.
//  K1: expand relative-position bias  bias[h][i][j] = table[rel[i*N+j]][h]
//  K2: qkv = x @ W_qkv  (bf16 WMMA, scatter to q/k/v [B][H][N][D] bf16)
//  K3: flash attention per (b,h): S=QK^T (WMMA) +scale+bias, online softmax,
//      O += P@V (WMMA), normalized O -> AO [B][N][H*D] bf16
//  K4: out = AO @ W_out + b_out  (bf16 WMMA, f32 out)
// ---------------------------------------------------------------------------

typedef __bf16 bf16;
typedef bf16  v16bf  __attribute__((ext_vector_type(16)));
typedef bf16  bf16x2 __attribute__((ext_vector_type(2)));
typedef float v8f    __attribute__((ext_vector_type(8)));
typedef unsigned int   u32x4 __attribute__((ext_vector_type(4)));
typedef unsigned int   u32x2 __attribute__((ext_vector_type(2)));
typedef unsigned short u16x4 __attribute__((ext_vector_type(4)));

#define B_  32
#define N_  1024
#define H_  8
#define D_  32
#define INNER_ 256
#define LDA 40            // padded LDS row stride (bf16 elements), 80B rows

struct B32x8 { u32x4 a, b; };

static __device__ inline v16bf frag2(u32x4 a, u32x4 b) {
  B32x8 t; t.a = a; t.b = b;
  return __builtin_bit_cast(v16bf, t);
}

static __device__ inline v8f zero8() {
  v8f z;
#pragma unroll
  for (int i = 0; i < 8; ++i) z[i] = 0.0f;
  return z;
}

// A fragment (16x32 bf16, MxK): lane l holds row (l&15); K elems are two
// contiguous 8-element chunks at K = hl*8 and K = 16 + hl*8  (ISA 7.12.2).
static __device__ inline v16bf load_a_frag(const bf16* base, int row, int stride, int hl) {
  const bf16* p = base + (size_t)row * stride + hl * 8;
  u32x4 lo = *(const u32x4*)(p);
  u32x4 hi = *(const u32x4*)(p + 16);
  return frag2(lo, hi);
}

// B fragment (32x16 bf16, KxN) from LDS stored [n][k]: lane l holds col (l&15),
// K elems are two 8-element chunks at K = hl*8 and K = 16 + hl*8 (matches A).
static __device__ inline v16bf load_b_frag(const bf16* base, int n, int stride, int hl) {
  const bf16* p = base + (size_t)n * stride + hl * 8;
  u32x4 lo = *(const u32x4*)(p);
  u32x4 hi = *(const u32x4*)(p + 16);
  return frag2(lo, hi);
}

static __device__ inline v8f wmma_bf16(v16bf a, v16bf b, v8f c) {
  return __builtin_amdgcn_wmma_f32_16x16x32_bf16(false, a, false, b, (short)0, c,
                                                 false, false);
}

// ---------------------------------------------------------------------------
// K1: bias expand  (N*N threads; bias fits in 192MB L2 afterwards)
// ---------------------------------------------------------------------------
__global__ __launch_bounds__(256) void k_bias_expand(
    const float* __restrict__ table, const int* __restrict__ rel,
    float* __restrict__ bias) {
  int ij = blockIdx.x * 256 + threadIdx.x;          // [0, N*N)
  int idx = rel[ij];
  const float* src = table + (size_t)idx * H_;
#pragma unroll
  for (int h = 0; h < H_; ++h)
    bias[(size_t)h * N_ * N_ + ij] = src[h];
}

// ---------------------------------------------------------------------------
// K2: qkv GEMM  M=32768 K=256 N=768, BM=128 BN=128 BK=32, 8 waves/block.
// Each wave: one A fragment x 8 B fragments -> 8 back-to-back WMMAs.
// ---------------------------------------------------------------------------
__global__ __launch_bounds__(256) void k_qkv_gemm(
    const float* __restrict__ x, const float* __restrict__ Wqkv,
    bf16* __restrict__ qw, bf16* __restrict__ kw, bf16* __restrict__ vw) {
  __shared__ __attribute__((aligned(32))) bf16 As[128 * LDA];
  __shared__ __attribute__((aligned(32))) bf16 Bs[128 * LDA];
  const int t = threadIdx.x, lane = t & 31, w = t >> 5;
  const int hl = lane >> 4, l15 = lane & 15;
  const int rbase = blockIdx.x * 128, nbase = blockIdx.y * 128;

  v8f acc[8];
#pragma unroll
  for (int c = 0; c < 8; ++c) acc[c] = zero8();

  for (int kk = 0; kk < 256; kk += 32) {
    __syncthreads();
    {   // stage A: 128x32 f32 -> bf16, packed b32 LDS stores
      int flat = t * 16;
      int row = flat >> 5, col = flat & 31;
      const float* src = x + (size_t)(rbase + row) * 256 + kk + col;
      bf16* dst = As + row * LDA + col;
#pragma unroll
      for (int i = 0; i < 8; ++i) {
        bf16x2 p;
        p.x = (bf16)src[2 * i];
        p.y = (bf16)src[2 * i + 1];
        *(bf16x2*)(dst + 2 * i) = p;
      }
    }
    {   // stage B transposed: Wqkv tile [32][128] -> Bs[n][k] bf16
      int flat = t * 16;
      int krow = flat >> 7, n0 = flat & 127;
      const float* src = Wqkv + (size_t)(kk + krow) * 768 + nbase + n0;
#pragma unroll
      for (int i = 0; i < 16; ++i) Bs[(n0 + i) * LDA + krow] = (bf16)src[i];
    }
    __syncthreads();
    v16bf af = load_a_frag(As, w * 16 + l15, LDA, hl);
    v16bf bfr[8];
#pragma unroll
    for (int c = 0; c < 8; ++c) bfr[c] = load_b_frag(Bs, c * 16 + l15, LDA, hl);
#pragma unroll
    for (int c = 0; c < 8; ++c) acc[c] = wmma_bf16(af, bfr[c], acc[c]);
  }
  // scatter C (f32) to q/k/v as bf16 [B][H][N][D]
  int rowb = rbase + w * 16 + hl * 8;
#pragma unroll
  for (int c = 0; c < 8; ++c) {
    int col = nbase + c * 16 + l15;
    int sec = col >> 8, inner = col & 255;
    int hh = inner >> 5, dd = inner & 31;
    bf16* dst = (sec == 0) ? qw : (sec == 1) ? kw : vw;
#pragma unroll
    for (int r = 0; r < 8; ++r) {
      int row = rowb + r;
      int b = row >> 10, n = row & 1023;
      dst[(((size_t)(b * H_ + hh)) * N_ + n) * D_ + dd] = (bf16)acc[c][r];
    }
  }
}

// ---------------------------------------------------------------------------
// K3: flash attention.  grid = (N/128, B*H), 8 waves, wave = 16 q-rows.
// ---------------------------------------------------------------------------
__global__ __launch_bounds__(256) void k_attn(
    const bf16* __restrict__ qw, const bf16* __restrict__ kw,
    const bf16* __restrict__ vw, const float* __restrict__ bias,
    bf16* __restrict__ AO) {
  __shared__ __attribute__((aligned(32))) bf16 Ks[32 * LDA];   // [key][dim]
  __shared__ __attribute__((aligned(32))) bf16 Vt[32 * LDA];   // [dim][key]
  __shared__ __attribute__((aligned(32))) bf16 Ps[8 * 16 * LDA];

  const int t = threadIdx.x, lane = t & 31, w = t >> 5;
  const int hl = lane >> 4, l15 = lane & 15;
  const int bh = blockIdx.y, b = bh >> 3, h = bh & 7;
  const bf16* qp = qw + (size_t)bh * N_ * D_;
  const bf16* kp = kw + (size_t)bh * N_ * D_;
  const bf16* vp = vw + (size_t)bh * N_ * D_;
  const float* bias_h = bias + (size_t)h * N_ * N_;
  const int qabs = blockIdx.x * 128 + w * 16;
  const float scale = 0.17677669529663687f;   // 1/sqrt(32)

  v16bf qf = load_a_frag(qp, qabs + l15, D_, hl);   // Q tile lives in regs
  v8f O0 = zero8(), O1 = zero8();
  float mr[8], lr[8];
#pragma unroll
  for (int r = 0; r < 8; ++r) { mr[r] = -__builtin_inff(); lr[r] = 0.0f; }

  const int keyrow = t >> 3, dg = (t & 7) * 4;      // staging mapping

  for (int j = 0; j < N_; j += 32) {
    __syncthreads();
    {   // stage K (direct) and V (transposed) tiles, 32 keys x 32 dims bf16
      u32x2 kv = *(const u32x2*)(kp + (size_t)(j + keyrow) * D_ + dg);
      *(u32x2*)(Ks + keyrow * LDA + dg) = kv;
      u32x2 vv = *(const u32x2*)(vp + (size_t)(j + keyrow) * D_ + dg);
      u16x4 vs = __builtin_bit_cast(u16x4, vv);
      Vt[(dg + 0) * LDA + keyrow] = __builtin_bit_cast(bf16, (unsigned short)vs.x);
      Vt[(dg + 1) * LDA + keyrow] = __builtin_bit_cast(bf16, (unsigned short)vs.y);
      Vt[(dg + 2) * LDA + keyrow] = __builtin_bit_cast(bf16, (unsigned short)vs.z);
      Vt[(dg + 3) * LDA + keyrow] = __builtin_bit_cast(bf16, (unsigned short)vs.w);
    }
    __syncthreads();

    // S = Q @ K^T : two 16x16 tiles covering 32 keys (fragments preloaded)
    v16bf kf0 = load_b_frag(Ks, l15, LDA, hl);
    v16bf kf1 = load_b_frag(Ks, 16 + l15, LDA, hl);
    v8f s0 = wmma_bf16(qf, kf0, zero8());
    v8f s1 = wmma_bf16(qf, kf1, zero8());

    // scale + relative position bias (L2-resident)
    float p0[8], p1[8];
    const int qrow0 = qabs + hl * 8;
#pragma unroll
    for (int r = 0; r < 8; ++r) {
      const float* bp = bias_h + (size_t)(qrow0 + r) * N_ + j + l15;
      p0[r] = s0[r] * scale + bp[0];
      p1[r] = s1[r] * scale + bp[16];
    }
    // online softmax: C-layout rows stripe by lane-half, so xor masks 1..8
    // reduce exactly one row per (VGPR slot, half).
#pragma unroll
    for (int r = 0; r < 8; ++r) {
      float v = fmaxf(p0[r], p1[r]);
      v = fmaxf(v, __shfl_xor(v, 1));
      v = fmaxf(v, __shfl_xor(v, 2));
      v = fmaxf(v, __shfl_xor(v, 4));
      v = fmaxf(v, __shfl_xor(v, 8));
      float mn = fmaxf(mr[r], v);
      float alpha = __expf(mr[r] - mn);
      mr[r] = mn;
      p0[r] = __expf(p0[r] - mn);
      p1[r] = __expf(p1[r] - mn);
      float rs = p0[r] + p1[r];
      rs += __shfl_xor(rs, 1);
      rs += __shfl_xor(rs, 2);
      rs += __shfl_xor(rs, 4);
      rs += __shfl_xor(rs, 8);
      lr[r] = lr[r] * alpha + rs;
      O0[r] *= alpha;
      O1[r] *= alpha;
    }
    // C-layout -> A-fragment transpose bounce through LDS (per-wave region)
    bf16* pw = Ps + w * 16 * LDA;
#pragma unroll
    for (int r = 0; r < 8; ++r) {
      pw[(hl * 8 + r) * LDA + l15]      = (bf16)p0[r];
      pw[(hl * 8 + r) * LDA + 16 + l15] = (bf16)p1[r];
    }
    __syncthreads();
    v16bf pf  = load_a_frag(pw, l15, LDA, hl);
    v16bf vf0 = load_b_frag(Vt, l15, LDA, hl);
    v16bf vf1 = load_b_frag(Vt, 16 + l15, LDA, hl);
    O0 = wmma_bf16(pf, vf0, O0);
    O1 = wmma_bf16(pf, vf1, O1);
  }

  // normalize and emit AO[b][q][h*32 + d] as bf16
#pragma unroll
  for (int r = 0; r < 8; ++r) {
    float inv = 1.0f / lr[r];
    int q = qabs + hl * 8 + r;
    size_t base = ((size_t)(b * N_ + q)) * INNER_ + h * D_;
    AO[base + l15]      = (bf16)(O0[r] * inv);
    AO[base + 16 + l15] = (bf16)(O1[r] * inv);
  }
}

// ---------------------------------------------------------------------------
// K4: output GEMM  M=32768 K=256 N=256 (+ bias), BM=128 BN=128, f32 out
// ---------------------------------------------------------------------------
__global__ __launch_bounds__(256) void k_out_gemm(
    const bf16* __restrict__ AO, const float* __restrict__ Wout,
    const float* __restrict__ bout, float* __restrict__ out) {
  __shared__ __attribute__((aligned(32))) bf16 As[128 * LDA];
  __shared__ __attribute__((aligned(32))) bf16 Bs[128 * LDA];
  const int t = threadIdx.x, lane = t & 31, w = t >> 5;
  const int hl = lane >> 4, l15 = lane & 15;
  const int rbase = blockIdx.x * 128, nbase = blockIdx.y * 128;

  v8f acc[8];
#pragma unroll
  for (int c = 0; c < 8; ++c) acc[c] = zero8();

  for (int kk = 0; kk < 256; kk += 32) {
    __syncthreads();
    {   // stage A: bf16 copy, 16 elements (two b128) per thread
      int row = t >> 1, ch = (t & 1) * 16;
      const u32x4* src = (const u32x4*)(AO + (size_t)(rbase + row) * 256 + kk + ch);
      u32x4 v0 = src[0], v1 = src[1];
      *(u32x4*)(As + row * LDA + ch)     = v0;
      *(u32x4*)(As + row * LDA + ch + 8) = v1;
    }
    {   // stage B transposed: Wout tile [32][128] -> Bs[n][k] bf16
      int flat = t * 16;
      int krow = flat >> 7, n0 = flat & 127;
      const float* src = Wout + (size_t)(kk + krow) * 256 + nbase + n0;
#pragma unroll
      for (int i = 0; i < 16; ++i) Bs[(n0 + i) * LDA + krow] = (bf16)src[i];
    }
    __syncthreads();
    v16bf af = load_a_frag(As, w * 16 + l15, LDA, hl);
    v16bf bfr[8];
#pragma unroll
    for (int c = 0; c < 8; ++c) bfr[c] = load_b_frag(Bs, c * 16 + l15, LDA, hl);
#pragma unroll
    for (int c = 0; c < 8; ++c) acc[c] = wmma_bf16(af, bfr[c], acc[c]);
  }
  int rowb = rbase + w * 16 + hl * 8;
#pragma unroll
  for (int c = 0; c < 8; ++c) {
    int col = nbase + c * 16 + l15;
    float bb = bout[col];
#pragma unroll
    for (int r = 0; r < 8; ++r)
      out[(size_t)(rowb + r) * 256 + col] = acc[c][r] + bb;
  }
}

// ---------------------------------------------------------------------------
extern "C" void kernel_launch(void* const* d_in, const int* in_sizes, int n_in,
                              void* d_out, int out_size, void* d_ws, size_t ws_size,
                              hipStream_t stream) {
  const float* x     = (const float*)d_in[0];
  const float* Wqkv  = (const float*)d_in[1];
  const float* table = (const float*)d_in[2];
  const float* Wout  = (const float*)d_in[3];
  const float* bout  = (const float*)d_in[4];
  const int*   rel   = (const int*)d_in[5];
  float* out = (float*)d_out;

  char* ws = (char*)d_ws;
  bf16*  qw   = (bf16*)(ws);                           // 16 MB
  bf16*  kw   = (bf16*)(ws + ((size_t)16 << 20));      // 16 MB
  bf16*  vw   = (bf16*)(ws + ((size_t)32 << 20));      // 16 MB
  float* bias = (float*)(ws + ((size_t)48 << 20));     // 32 MB
  bf16*  AO   = (bf16*)(ws + ((size_t)80 << 20));      // 16 MB

  k_bias_expand<<<dim3(N_ * N_ / 256), 256, 0, stream>>>(table, rel, bias);
  k_qkv_gemm<<<dim3((B_ * N_) / 128, 768 / 128), 256, 0, stream>>>(x, Wqkv, qw, kw, vw);
  k_attn<<<dim3(N_ / 128, B_ * H_), 256, 0, stream>>>(qw, kw, vw, bias, AO);
  k_out_gemm<<<dim3((B_ * N_) / 128, 256 / 128), 256, 0, stream>>>(AO, Wout, bout, out);
}
